// MHA_8813272891727
// MI455X (gfx1250) — matrix operand order repaired and checked by reading, not compiled, not run on hardware
//
#include <hip/hip_runtime.h>
#include <hip/hip_bf16.h>

// Problem constants (match reference)
#define TOKEN_DIM 1024
#define N_HEADS   16
#define HEAD_DIM  64
#define BATCH     8
#define SEQ       1024

// 1/sqrt(64) * log2(e): flash softmax runs in the log2 domain (exp2 only).
#define QSCALE 0.18033688011112042591999058512524f

typedef _Float16 v16h __attribute__((ext_vector_type(16)));
typedef _Float16 v8h  __attribute__((ext_vector_type(8)));
typedef float    v8f  __attribute__((ext_vector_type(8)));

static __device__ __forceinline__ v8f wmma_f16(v16h a, v16h b, v8f c) {
  // D = A(16x32 f16) * B(32x16 f16) + C(16x16 f32)
  return __builtin_amdgcn_wmma_f32_16x16x32_f16(
      /*neg_a=*/false, a, /*neg_b=*/false, b,
      /*c_mod=*/(short)0, c, /*reuse_a=*/false, /*reuse_b=*/false);
}

// bare v_exp_f32 (exp2)
static __device__ __forceinline__ float exp2_fast(float x) {
  return __builtin_amdgcn_exp2f(x);
}

// ---------------------------------------------------------------------------
// Kernel 0: pack Wq/Wk/Wv into per-lane WMMA B-layout, f32 -> f16.
// Layout: Wpk[w][h][nt][ch][lane][j] (j = 0..15 contiguous f16 = one 32B load)
//   value = W_w[h][ ch*32 + (lane>>4)*16 + j ][ nt*16 + (lane&15) ]
// ---------------------------------------------------------------------------
__global__ __launch_bounds__(256) void mha_pack_w_kernel(
    const float* __restrict__ Wq, const float* __restrict__ Wk,
    const float* __restrict__ Wv, _Float16* __restrict__ Wpk)
{
  const int t = blockIdx.x * blockDim.x + threadIdx.x;  // 0..12287
  if (t >= 3 * 16 * 4 * 2 * 32) return;
  const int lane = t & 31;
  const int ch   = (t >> 5) & 1;
  const int nt   = (t >> 6) & 3;
  const int h    = (t >> 8) & 15;
  const int w    = t >> 12;
  const float* W = (w == 0) ? Wq : (w == 1) ? Wk : Wv;
  const float* Wh = W + h * 4096;
  const int n  = nt * 16 + (lane & 15);
  const int kb = ch * 32 + (lane >> 4) * 16;
  v16h v;
#pragma unroll
  for (int j = 0; j < 16; ++j) v[j] = (_Float16)Wh[(size_t)(kb + j) * 64 + n];
  *(v16h*)(Wpk + (size_t)t * 16) = v;
}

// ---------------------------------------------------------------------------
// Kernel 1: per-head Q/K/V projections with WMMA (B operands from packed W).
// C operand is inline 0 (no splat movs); bias is folded into the stores.
//   Q stored [B*H, S, 64] f16, pre-scaled by log2(e)/sqrt(64)
//   K stored [B*H, S, 64] f16
//   Vt stored [B*H, 64, S] f16 (transposed)
// One wave per 16-row tile of one (b,h).
// ---------------------------------------------------------------------------
__global__ __launch_bounds__(128) void mha_proj_kernel(
    const float* __restrict__ x,
    const _Float16* __restrict__ Wpk,
    const float* __restrict__ bq, const float* __restrict__ bk,
    const float* __restrict__ bv,
    _Float16* __restrict__ Q, _Float16* __restrict__ K,
    _Float16* __restrict__ Vt)
{
  const int lane = threadIdx.x & 31;
  const int wave = threadIdx.x >> 5;
  const int tile = blockIdx.x * 4 + wave;       // 0..8191
  const int bh   = tile >> 6;                   // b*16 + h
  const int mt   = tile & 63;
  const int h    = bh & 15;
  const int b    = bh >> 4;
  const int s0   = mt * 16;
  const int l16  = lane & 15;
  const int hsel = lane >> 4;

  // ---- X tile (16 x 64) -> WMMA A layout (two 16x32 f16 operands) ----
  const float* xrow =
      x + (size_t)(b * SEQ + s0 + l16) * TOKEN_DIM + h * HEAD_DIM;
  v16h a0, a1;
  {
    const int kb = hsel * 16;
#pragma unroll
    for (int c = 0; c < 2; ++c) {
      v8f lo = *(const v8f*)(xrow + c * 32 + kb);
      v8f hi = *(const v8f*)(xrow + c * 32 + kb + 8);
      v16h a;
#pragma unroll
      for (int j = 0; j < 8; ++j) { a[j] = (_Float16)lo[j]; a[j + 8] = (_Float16)hi[j]; }
      if (c == 0) a0 = a; else a1 = a;
    }
  }

  const float* biash[3] = {bq + h * 64, bk + h * 64, bv + h * 64};
  const v8f zero = {};

#pragma unroll
  for (int o = 0; o < 3; ++o) {
    v8f acc[4];
    float bval[4];
#pragma unroll
    for (int nt = 0; nt < 4; ++nt) {
      bval[nt] = biash[o][nt * 16 + l16];        // per-column scalar
      // packed B operands: one 32B contiguous load each
      const _Float16* wp =
          Wpk + ((size_t)((((o * 16 + h) * 4 + nt) * 2) * 32) + lane) * 16;
      v16h b0 = *(const v16h*)(wp);
      v16h b1 = *(const v16h*)(wp + 32 * 16);
      acc[nt] = wmma_f16(a0, b0, zero);          // SRC2 = inline 0
      acc[nt] = wmma_f16(a1, b1, acc[nt]);
    }
    // C layout: VGPR r, lane -> row = r + hsel*8, col = l16
    if (o == 0) {
      _Float16* Qb = Q + (size_t)bh * SEQ * HEAD_DIM;
#pragma unroll
      for (int nt = 0; nt < 4; ++nt) {
        const float bs = bval[nt] * QSCALE;
#pragma unroll
        for (int r = 0; r < 8; ++r)              // (acc + b) * QSCALE, one fma
          Qb[(size_t)(s0 + r + hsel * 8) * HEAD_DIM + nt * 16 + l16] =
              (_Float16)fmaf(acc[nt][r], QSCALE, bs);
      }
    } else if (o == 1) {
      _Float16* Kb = K + (size_t)bh * SEQ * HEAD_DIM;
#pragma unroll
      for (int nt = 0; nt < 4; ++nt)
#pragma unroll
        for (int r = 0; r < 8; ++r)
          Kb[(size_t)(s0 + r + hsel * 8) * HEAD_DIM + nt * 16 + l16] =
              (_Float16)(acc[nt][r] + bval[nt]);
    } else {
      _Float16* Vb = Vt + (size_t)bh * HEAD_DIM * SEQ;
#pragma unroll
      for (int nt = 0; nt < 4; ++nt) {
        v8h pack;
#pragma unroll
        for (int r = 0; r < 8; ++r) pack[r] = (_Float16)(acc[nt][r] + bval[nt]);
        *(v8h*)(Vb + (size_t)(nt * 16 + l16) * SEQ + s0 + hsel * 8) = pack;
      }
    }
  }
}

// ---------------------------------------------------------------------------
// Kernel 2: flash attention, transposed formulation (no LDS transpose).
//   S^T = K * Q^T   (A = K tile, B = Q^T)         -> softmax reduces in-register
//   O^T += V^T * P^T (A = V^T tile, B = P^T)      -> contiguous fp32 out stores
// Scores are already in the log2 domain (Q pre-scaled), so softmax uses bare
// v_exp_f32 (exp2) with no per-element multiplies.
// One wave per 32 query rows; K/V bytes shared by both 16-row m-tiles.
// ---------------------------------------------------------------------------
__global__ __launch_bounds__(128) void mha_flash_kernel(
    const _Float16* __restrict__ Q, const _Float16* __restrict__ K,
    const _Float16* __restrict__ Vt, float* __restrict__ out)
{
  const int lane = threadIdx.x & 31;
  const int wave = threadIdx.x >> 5;
  const int tile = blockIdx.x * 4 + wave;       // 0..4095
  const int bh   = tile >> 5;                   // 0..127
  const int mt   = tile & 31;                   // 32-row query tile
  const int b    = bh >> 4;
  const int h    = bh & 15;
  const int s0   = mt * 32;
  const int l16  = lane & 15;
  const int hsel = lane >> 4;

  // ---- Q^T as WMMA B operands (resident): Qb[mi][chunk] ----
  // B layout: lane n = m = l16; lanes0-15 K=d 0..15, lanes16-31 K=d 16..31
  v16h Qb[2][2];
#pragma unroll
  for (int mi = 0; mi < 2; ++mi)
#pragma unroll
    for (int c = 0; c < 2; ++c)
      Qb[mi][c] = *(const v16h*)(Q +
          (size_t)(bh * SEQ + s0 + mi * 16 + l16) * HEAD_DIM + c * 32 + hsel * 16);

  v8f o[2][4];
#pragma unroll
  for (int mi = 0; mi < 2; ++mi)
#pragma unroll
    for (int dt = 0; dt < 4; ++dt) o[mi][dt] = (v8f){};
  float mrun[2] = {-1e30f, -1e30f};
  float lrun[2] = {0.0f, 0.0f};

  const _Float16* Kbh = K  + (size_t)bh * SEQ * HEAD_DIM;
  const _Float16* Vbh = Vt + (size_t)bh * HEAD_DIM * SEQ;

#pragma unroll 1
  for (int kt = 0; kt < SEQ / 32; ++kt) {
    const int t0 = kt * 32;

    // ---- K tile as A operands: Ka[sub][chunk], rows t = t0+sub*16+l16 ----
    v16h Ka[2][2];
#pragma unroll
    for (int sub = 0; sub < 2; ++sub) {
      const _Float16* Kr = Kbh + (size_t)(t0 + sub * 16 + l16) * HEAD_DIM;
#pragma unroll
      for (int c = 0; c < 2; ++c) {
        v8h lo = *(const v8h*)(Kr + c * 32 + hsel * 16);
        v8h hi = *(const v8h*)(Kr + c * 32 + hsel * 16 + 8);
        v16h a;
#pragma unroll
        for (int j = 0; j < 8; ++j) { a[j] = lo[j]; a[j + 8] = hi[j]; }
        Ka[sub][c] = a;
      }
    }
    if (kt + 1 < SEQ / 32) {  // prefetch next chunk (global_prefetch_b8)
      __builtin_prefetch(Kbh + (size_t)(t0 + 32 + l16) * HEAD_DIM, 0, 1);
      __builtin_prefetch(Kbh + (size_t)(t0 + 48 + l16) * HEAD_DIM, 0, 1);
      __builtin_prefetch(Vbh + (size_t)lane * SEQ + t0 + 32, 0, 1);
      __builtin_prefetch(Vbh + (size_t)(32 + lane) * SEQ + t0 + 32, 0, 1);
    }

    // ---- S^T tiles: rows t, cols m.  8 WMMAs ----
    v8f sT[2][2];
    {
      v8f z = {};
#pragma unroll
      for (int mi = 0; mi < 2; ++mi)
#pragma unroll
        for (int sub = 0; sub < 2; ++sub) {
          v8f s = wmma_f16(Ka[sub][0], Qb[mi][0], z);
          sT[mi][sub] = wmma_f16(Ka[sub][1], Qb[mi][1], s);
        }
    }

    // ---- V^T tiles as A operands: Va[dt], rows d = dt*16+l16, K = t ----
    v16h Va[4];
#pragma unroll
    for (int dt = 0; dt < 4; ++dt) {
      const _Float16* Vr = Vbh + (size_t)(dt * 16 + l16) * SEQ + t0 + hsel * 16;
      v8h lo = *(const v8h*)(Vr);
      v8h hi = *(const v8h*)(Vr + 8);
      v16h a;
#pragma unroll
      for (int j = 0; j < 8; ++j) { a[j] = lo[j]; a[j + 8] = hi[j]; }
      Va[dt] = a;
    }

    // ---- per m-tile: online softmax over t (in-register) + P^T build + PV ----
#pragma unroll
    for (int mi = 0; mi < 2; ++mi) {
      // lane column m = l16; tile a: t = r + hsel*8; tile b: t = 16 + r + hsel*8
      float rm = -1e30f;
#pragma unroll
      for (int r = 0; r < 8; ++r)
        rm = fmaxf(rm, fmaxf(sT[mi][0][r], sT[mi][1][r]));
      rm = fmaxf(rm, __shfl_xor(rm, 16, 32));     // join the two half-rows
      const float mnew = fmaxf(mrun[mi], rm);
      const float corr = exp2_fast(mrun[mi] - mnew);  // log2-domain: bare v_exp
      float pa[8], pb[8], rs = 0.0f;
#pragma unroll
      for (int r = 0; r < 8; ++r) {
        pa[r] = exp2_fast(sT[mi][0][r] - mnew);
        pb[r] = exp2_fast(sT[mi][1][r] - mnew);
        rs += pa[r] + pb[r];
      }
      rs += __shfl_xor(rs, 16, 32);
      lrun[mi] = lrun[mi] * corr + rs;
      mrun[mi] = mnew;
#pragma unroll
      for (int dt = 0; dt < 4; ++dt)
#pragma unroll
        for (int i = 0; i < 8; ++i) o[mi][dt][i] *= corr;

      // Build P^T as B operand: lanes0-15 K=t 0..15, lanes16-31 K=t 16..31.
      // Exchange 8 f16 (4 dwords) with partner lane (xor 16).
      union { _Float16 hh[8]; int ii[4]; } snd, rcv;
#pragma unroll
      for (int r = 0; r < 8; ++r)
        snd.hh[r] = hsel ? (_Float16)pa[r] : (_Float16)pb[r];
#pragma unroll
      for (int i = 0; i < 4; ++i) rcv.ii[i] = __shfl_xor(snd.ii[i], 16, 32);
      v16h Pb;
#pragma unroll
      for (int j = 0; j < 8; ++j) {
        Pb[j]     = hsel ? rcv.hh[j] : (_Float16)pa[j];
        Pb[8 + j] = hsel ? (_Float16)pb[j] : rcv.hh[j];
      }

      // O^T += V^T x P^T  (K = 32 keys).  4 WMMAs per m-tile.
#pragma unroll
      for (int dt = 0; dt < 4; ++dt)
        o[mi][dt] = wmma_f16(Va[dt], Pb, o[mi][dt]);
    }
  }

  // ---- finalize: O^T C-layout -> contiguous 32B fp32 stores ----
#pragma unroll
  for (int mi = 0; mi < 2; ++mi) {
    const float inv = 1.0f / lrun[mi];
    const int row = s0 + mi * 16 + l16;           // query row = lane column
    float* orow = out + (size_t)(b * SEQ + row) * TOKEN_DIM + h * HEAD_DIM;
#pragma unroll
    for (int dt = 0; dt < 4; ++dt) {
      v8f v;
#pragma unroll
      for (int i = 0; i < 8; ++i) v[i] = o[mi][dt][i] * inv;
      *(v8f*)(orow + dt * 16 + hsel * 8) = v;     // d = dt*16 + hsel*8 + r
    }
  }
}

extern "C" void kernel_launch(void* const* d_in, const int* in_sizes, int n_in,
                              void* d_out, int out_size, void* d_ws, size_t ws_size,
                              hipStream_t stream) {
  const float* x  = (const float*)d_in[0];
  const float* Wq = (const float*)d_in[1];
  const float* Wk = (const float*)d_in[2];
  const float* Wv = (const float*)d_in[3];
  const float* bq = (const float*)d_in[4];
  const float* bk = (const float*)d_in[5];
  const float* bv = (const float*)d_in[6];
  float* out = (float*)d_out;

  const size_t elems = (size_t)BATCH * N_HEADS * SEQ * HEAD_DIM;  // 8,388,608
  _Float16* Qw  = (_Float16*)d_ws;         // [B*H, S, 64]  f16 (16 MB)
  _Float16* Kw  = Qw + elems;              // [B*H, S, 64]  f16 (16 MB)
  _Float16* Vw  = Kw + elems;              // [B*H, 64, S]  f16 (16 MB)
  _Float16* Wpk = Vw + elems;              // packed weights (384 KB)

  mha_pack_w_kernel<<<48, 256, 0, stream>>>(Wq, Wk, Wv, Wpk);
  mha_proj_kernel<<<2048, 128, 0, stream>>>(x, Wpk, bq, bk, bv, Qw, Kw, Vw);
  mha_flash_kernel<<<1024, 128, 0, stream>>>(Qw, Kw, Vw, out);

  (void)in_sizes; (void)n_in; (void)out_size; (void)ws_size;
}
